// _PointFeatureProjector_65781719106307
// MI455X (gfx1250) — matrix operand order repaired and checked by reading, not compiled, not run on hardware
//
#include <hip/hip_runtime.h>
#include <hip/hip_fp16.h>

typedef __attribute__((ext_vector_type(4)))  float    v4f;
typedef __attribute__((ext_vector_type(8)))  float    v8f;
typedef __attribute__((ext_vector_type(8)))  _Float16 v8h;
typedef __attribute__((ext_vector_type(16))) _Float16 v16h;

#define D_MODEL      512
#define NPTS         65536
#define BATCH        8
#define STATE_DIM    16
#define MASK_BUCKETS 2048
#define PPB          32   // points per block in the bulk kernel

// ---------------------------------------------------------------------------
// Bulk kernel: h[b,n,:] = xyz@Wxyz + ra*rgb@Wrgb + ga*gfeat@Wg + mask_embed[m]
// HBM-bound (1.07 GB stores). 128 threads/block; thread owns 4 consecutive d
// columns (float4). All 10 weight columns live in registers; per point we do
// one L2-resident gather + 40 FMAs + one 128-bit non-temporal store.
// ---------------------------------------------------------------------------
__global__ __launch_bounds__(128) void point_proj_kernel(
    const float* __restrict__ xyz,   const float* __restrict__ state,
    const float* __restrict__ rgb,   const unsigned int* __restrict__ mask_id,
    const float* __restrict__ Wxyz,  const float* __restrict__ Wrgb,
    const float* __restrict__ rgb_alpha, const float* __restrict__ Wg,
    const float* __restrict__ grip_alpha, const float* __restrict__ mask_embed,
    float* __restrict__ h)
{
    const int d = threadIdx.x * 4;              // this thread's 4 d-columns

    // weight slices for these 4 columns (register-resident, reused PPB times)
    const v4f wx0 = *(const v4f*)(Wxyz + 0 * D_MODEL + d);
    const v4f wx1 = *(const v4f*)(Wxyz + 1 * D_MODEL + d);
    const v4f wx2 = *(const v4f*)(Wxyz + 2 * D_MODEL + d);
    const v4f wr0 = *(const v4f*)(Wrgb + 0 * D_MODEL + d);
    const v4f wr1 = *(const v4f*)(Wrgb + 1 * D_MODEL + d);
    const v4f wr2 = *(const v4f*)(Wrgb + 2 * D_MODEL + d);
    const v4f wg0 = *(const v4f*)(Wg   + 0 * D_MODEL + d);
    const v4f wg1 = *(const v4f*)(Wg   + 1 * D_MODEL + d);
    const v4f wg2 = *(const v4f*)(Wg   + 2 * D_MODEL + d);
    const v4f wg3 = *(const v4f*)(Wg   + 3 * D_MODEL + d);

    const float ra = rgb_alpha[0];
    const float ga = grip_alpha[0];

    const int base = blockIdx.x * PPB;          // PPB divides NPTS -> b uniform
    const int b    = base / NPTS;
    const float gx = state[b * STATE_DIM + 0];
    const float gy = state[b * STATE_DIM + 1];
    const float gz = state[b * STATE_DIM + 2];

    for (int p = 0; p < PPB; ++p) {
        const int idx = base + p;               // flat (b,n) point index
        const float x0 = xyz[idx * 3 + 0];
        const float x1 = xyz[idx * 3 + 1];
        const float x2 = xyz[idx * 3 + 2];
        const float r0 = rgb[idx * 3 + 0];
        const float r1 = rgb[idx * 3 + 1];
        const float r2 = rgb[idx * 3 + 2];
        const unsigned m = mask_id[idx] & (MASK_BUCKETS - 1);

        const float e0 = x0 - gx, e1 = x1 - gy, e2 = x2 - gz;
        const float dist = sqrtf(e0 * e0 + e1 * e1 + e2 * e2);

        // gather hits the 4 MB table -> L2 resident
        v4f acc = *(const v4f*)(mask_embed + (size_t)m * D_MODEL + d);

        acc += x0 * wx0 + x1 * wx1 + x2 * wx2;
        acc += ra * (r0 * wr0 + r1 * wr1 + r2 * wr2);
        acc += ga * (e0 * wg0 + e1 * wg1 + e2 * wg2 + dist * wg3);

        // NT store: don't let the 1 GB write stream evict the embed table
        __builtin_nontemporal_store(acc,
            (v4f*)(h + (size_t)idx * D_MODEL + d));
    }
}

// ---------------------------------------------------------------------------
// Layer 1 + SiLU: act[b][d] = silu(state[b]@Ws1[:,d] + bs1[d]) as f16,
// padded to M=16 rows (rows 8..15 zero) for the WMMA A-matrix.
// ---------------------------------------------------------------------------
__global__ __launch_bounds__(512) void state_mlp_prep(
    const float* __restrict__ state, const float* __restrict__ Ws1,
    const float* __restrict__ bs1, _Float16* __restrict__ act)
{
    const int d = threadIdx.x;                  // 512 threads, one per column
    const float bias = bs1[d];
    for (int b = 0; b < BATCH; ++b) {
        float s = bias;
        #pragma unroll
        for (int k = 0; k < STATE_DIM; ++k)
            s += state[b * STATE_DIM + k] * Ws1[k * D_MODEL + d];
        const float silu = s / (1.0f + __expf(-s));
        act[b * D_MODEL + d] = (_Float16)silu;
    }
    for (int b = BATCH; b < 16; ++b)
        act[b * D_MODEL + d] = (_Float16)0.0f;
}

// ---------------------------------------------------------------------------
// Layer 2 via WMMA: st = act(16x512,f16) @ Ws2(512x512) + bs2, f32 accum.
// One wave per 16-wide N tile; K=512 -> 16 x v_wmma_f32_16x16x32_f16.
// A fragment (ISA 16-bit A layout): M = lane&15; lanes 0-15 take K 0-7/16-23,
// lanes 16-31 take K 8-15/24-31 -> two contiguous v8h loads per lane.
// B fragment: N = lane&15; same K order as A (lanes 0-15 K 0-7/16-23, 16-31 K 8-15/24-31) (f32->f16).
// ---------------------------------------------------------------------------
__global__ __launch_bounds__(32) void state_mlp_wmma(
    const _Float16* __restrict__ act, const float* __restrict__ Ws2,
    const float* __restrict__ bs2, float* __restrict__ st_out)
{
    const int nt    = blockIdx.x;               // 0..31 N tile
    const int lane  = threadIdx.x;              // 0..31
    const int M     = lane & 15;
    const int koffA = (lane < 16) ? 0 : 8;
    const int koffB = (lane < 16) ? 0 : 16;
    const int ncol  = nt * 16 + (lane & 15);

    v8f c = {};
    for (int kt = 0; kt < 16; ++kt) {
        const int kbase = kt * 32;

        const v8h lo = *(const v8h*)(act + M * D_MODEL + kbase + koffA);
        const v8h hi = *(const v8h*)(act + M * D_MODEL + kbase + 16 + koffA);
        v16h a;
        #pragma unroll
        for (int j = 0; j < 8; ++j) { a[j] = lo[j]; a[8 + j] = hi[j]; }

        v16h bm;
        #pragma unroll
        for (int j = 0; j < 16; ++j)
            bm[j] = (_Float16)Ws2[(size_t)(kbase + koffA + j + ((j & 8) ? 8 : 0)) * D_MODEL + ncol];

        // (neg_a, A, neg_b, B, c_mod, C, reuse_a, reuse_b)
        c = __builtin_amdgcn_wmma_f32_16x16x32_f16(
                false, a, false, bm, (short)0, c, false, false);
    }

    // C/D layout: VGPR r, lanes 0-15 hold M=r (only M<8 is real data)
    if (lane < 16) {
        const float bias = bs2[ncol];
        #pragma unroll
        for (int r = 0; r < 8; ++r)
            st_out[(size_t)r * D_MODEL + ncol] = c[r] + bias;
    }
}

extern "C" void kernel_launch(void* const* d_in, const int* in_sizes, int n_in,
                              void* d_out, int out_size, void* d_ws, size_t ws_size,
                              hipStream_t stream) {
    (void)in_sizes; (void)n_in; (void)out_size; (void)ws_size;

    const float*        xyz        = (const float*)d_in[0];
    const float*        state      = (const float*)d_in[1];
    const float*        rgb        = (const float*)d_in[2];
    const unsigned int* mask_id    = (const unsigned int*)d_in[3];
    const float*        Wxyz       = (const float*)d_in[4];
    const float*        Wrgb       = (const float*)d_in[5];
    const float*        rgb_alpha  = (const float*)d_in[6];
    const float*        Wg         = (const float*)d_in[7];
    const float*        grip_alpha = (const float*)d_in[8];
    const float*        Ws1        = (const float*)d_in[9];
    const float*        bs1        = (const float*)d_in[10];
    const float*        Ws2        = (const float*)d_in[11];
    const float*        bs2        = (const float*)d_in[12];
    const float*        mask_embed = (const float*)d_in[13];

    float* h_out  = (float*)d_out;
    float* st_out = h_out + (size_t)BATCH * NPTS * D_MODEL;   // tuple part 2
    _Float16* act = (_Float16*)d_ws;                          // 16x512 f16

    // bulk HBM-bound path: 16384 blocks x 128 threads, 32 points each
    point_proj_kernel<<<(BATCH * NPTS) / PPB, 128, 0, stream>>>(
        xyz, state, rgb, mask_id, Wxyz, Wrgb, rgb_alpha, Wg, grip_alpha,
        mask_embed, h_out);

    // tiny state MLP: prep (layer1+SiLU, f16 pack) then WMMA layer2
    state_mlp_prep<<<1, 512, 0, stream>>>(state, Ws1, bs1, act);
    state_mlp_wmma<<<32, 32, 0, stream>>>(act, Ws2, bs2, st_out);
}
